// TinyMNISTNet_88012469830412
// MI455X (gfx1250) — matrix operand order repaired and checked by reading, not compiled, not run on hardware
//
#include <hip/hip_runtime.h>
#include <hip/hip_bf16.h>

// ---------------------------------------------------------------------------
// TWN ternary MLP: out = relu(x @ Wq1^T + b1) @ Wq2^T + b2
//   x: [65536, 784] f32,  W1: [64,784], b1: [64], W2: [10,64], b2: [10]
// Memory-bound (AI ~32 FLOP/B << ~200 FLOP/B crossover at f16-WMMA rates).
// HBM floor: ~208 MB / 23.3 TB/s ~= 9 us. Compute via v_wmma_f32_16x16x32_f16
// with x split hi+lo f16 (near-f32 accuracy); ternary weights exact in f16.
// Quantization is grid-parallel (3 tiny kernels, deterministic two-level
// reductions via workspace partials) so it doesn't serialize on one WGP.
// ---------------------------------------------------------------------------

typedef __attribute__((ext_vector_type(16))) _Float16 v16h;
typedef __attribute__((ext_vector_type(8)))  _Float16 v8h;
typedef __attribute__((ext_vector_type(8)))  float    v8f;

#define K1_TILES 25               // 784 padded to 25 * 32 = 800
#define N1_TILES 4                // 64 hidden = 4 x 16
#define W1FRAG_HALVES (K1_TILES * N1_TILES * 32 * 16)   // 51200
#define W2FRAG_HALVES (2 * 32 * 16)                     // 1024 (K=64 -> 2 tiles)
#define NW1 (64 * 784)
#define NW2 (10 * 64)
#define QBLOCKS 64
#define QTHREADS 256

// ws float-partial slots (relative to ws+64, as float index):
//   [0..63]    p_abs1   [64..127]  p_abs2
//   [128..191] p_sm1    [192..255] p_cnt1
//   [256..319] p_sm2    [320..383] p_cnt2
#define P_ABS1 0
#define P_ABS2 64
#define P_SM1  128
#define P_CNT1 192
#define P_SM2  256
#define P_CNT2 320

__device__ __forceinline__ float block_reduce256(float v, float* sdata) {
    const int t = threadIdx.x;
    sdata[t] = v;
    __syncthreads();
    for (int s = 128; s > 0; s >>= 1) {
        if (t < s) sdata[t] += sdata[t + s];
        __syncthreads();
    }
    float r = sdata[0];
    __syncthreads();
    return r;
}

// ---- pass 1: per-block partial sum |W| for both tensors --------------------
__global__ __launch_bounds__(QTHREADS) void twn_abs_partial(
    const float* __restrict__ W1, const float* __restrict__ W2,
    float* __restrict__ part)
{
    __shared__ float sdata[QTHREADS];
    const int t = threadIdx.x, b = blockIdx.x;
    const int gid = b * QTHREADS + t, stride = QBLOCKS * QTHREADS;

    float s1 = 0.f, s2 = 0.f;
    for (int i = gid; i < NW1; i += stride) s1 += fabsf(W1[i]);
    for (int i = gid; i < NW2; i += stride) s2 += fabsf(W2[i]);

    s1 = block_reduce256(s1, sdata);
    s2 = block_reduce256(s2, sdata);
    if (t == 0) { part[P_ABS1 + b] = s1; part[P_ABS2 + b] = s2; }
}

// ---- pass 2: rebuild deltas from partials, emit masked-sum/count partials --
__global__ __launch_bounds__(QTHREADS) void twn_masked_partial(
    const float* __restrict__ W1, const float* __restrict__ W2,
    float* __restrict__ part)
{
    __shared__ float sdata[QTHREADS];
    const int t = threadIdx.x, b = blockIdx.x;
    const int gid = b * QTHREADS + t, stride = QBLOCKS * QTHREADS;

    float v1 = (t < QBLOCKS) ? part[P_ABS1 + t] : 0.f;
    float v2 = (t < QBLOCKS) ? part[P_ABS2 + t] : 0.f;
    const float delta1 = 0.7f * block_reduce256(v1, sdata) / (float)NW1;
    const float delta2 = 0.7f * block_reduce256(v2, sdata) / (float)NW2;

    float sm1 = 0.f, c1 = 0.f, sm2 = 0.f, c2 = 0.f;
    for (int i = gid; i < NW1; i += stride) {
        float a = fabsf(W1[i]);
        if (a > delta1) { sm1 += a; c1 += 1.f; }
    }
    for (int i = gid; i < NW2; i += stride) {
        float a = fabsf(W2[i]);
        if (a > delta2) { sm2 += a; c2 += 1.f; }
    }
    sm1 = block_reduce256(sm1, sdata);
    c1  = block_reduce256(c1,  sdata);
    sm2 = block_reduce256(sm2, sdata);
    c2  = block_reduce256(c2,  sdata);
    if (t == 0) {
        part[P_SM1 + b] = sm1; part[P_CNT1 + b] = c1;
        part[P_SM2 + b] = sm2; part[P_CNT2 + b] = c2;
    }
}

// ---- pass 3: alphas + write ternary weights in WMMA B-fragment layout ------
//   fragment (k_tile, n_tile): lane = (K%16>=8)*16 + (N%16), half = (K%32>=16)*8 + K%8
__global__ __launch_bounds__(QTHREADS) void twn_quant_write(
    const float* __restrict__ W1, const float* __restrict__ W2,
    _Float16* __restrict__ w1f, _Float16* __restrict__ w2f,
    const float* __restrict__ part, float* __restrict__ alphas)
{
    __shared__ float sdata[QTHREADS];
    const int t = threadIdx.x, b = blockIdx.x;
    const int gid = b * QTHREADS + t, stride = QBLOCKS * QTHREADS;

    float a1 = (t < QBLOCKS) ? part[P_ABS1 + t] : 0.f;
    float a2 = (t < QBLOCKS) ? part[P_ABS2 + t] : 0.f;
    float s1 = (t < QBLOCKS) ? part[P_SM1 + t]  : 0.f;
    float c1 = (t < QBLOCKS) ? part[P_CNT1 + t] : 0.f;
    float s2 = (t < QBLOCKS) ? part[P_SM2 + t]  : 0.f;
    float c2 = (t < QBLOCKS) ? part[P_CNT2 + t] : 0.f;

    const float delta1 = 0.7f * block_reduce256(a1, sdata) / (float)NW1;
    const float delta2 = 0.7f * block_reduce256(a2, sdata) / (float)NW2;
    const float alpha1 = block_reduce256(s1, sdata) / fmaxf(block_reduce256(c1, sdata), 1.f);
    const float alpha2 = block_reduce256(s2, sdata) / fmaxf(block_reduce256(c2, sdata), 1.f);

    if (b == 0 && t == 0) { alphas[0] = alpha1; alphas[1] = alpha2; }

    for (int idx = gid; idx < W1FRAG_HALVES; idx += stride) {
        int h     = idx & 15;
        int lane  = (idx >> 4) & 31;
        int ntile = (idx >> 9) & 3;
        int ktile = idx >> 11;
        int N = ntile * 16 + (lane & 15);
        int K = ktile * 32 + ((lane >= 16) ? 8 : 0) + (h & 7) + ((h >= 8) ? 16 : 0);
        float w = (K < 784) ? W1[N * 784 + K] : 0.f;
        float q = (fabsf(w) > delta1) ? ((w > 0.f) ? 1.f : -1.f) : 0.f;
        w1f[idx] = (_Float16)q;
    }
    for (int idx = gid; idx < W2FRAG_HALVES; idx += stride) {
        int h    = idx & 15;
        int lane = (idx >> 4) & 31;
        int k2   = idx >> 9;
        int N = lane & 15;
        int K = k2 * 32 + ((lane >= 16) ? 8 : 0) + (h & 7) + ((h >= 8) ? 16 : 0);
        float w = (N < 10 && K < 64) ? W2[N * 64 + K] : 0.f;
        float q = (fabsf(w) > delta2) ? ((w > 0.f) ? 1.f : -1.f) : 0.f;
        w2f[idx] = (_Float16)q;
    }
}

// ---------------------------------------------------------------------------
// Main fused MLP kernel. 256 threads = 8 waves; each wave computes a
// 16(M) x 64(N) hidden strip via 4 f32 accumulators, K-loop of 25 x 32,
// then the tiny second layer in-wave via LDS re-layout + 4 more WMMAs.
// A-fragment layout (16-bit A 16x32): lane<16 -> M=lane, K={0..7,16..23};
// lane>=16 -> M=lane-16, K={8..15,24..31}; half h -> K = (h/8)*16 + hsel*8 + h%8
// ---------------------------------------------------------------------------
__global__ __launch_bounds__(256) void twn_mlp_kernel(
    const float* __restrict__ x,
    const float* __restrict__ b1,
    const float* __restrict__ b2,
    const _Float16* __restrict__ w1f,
    const _Float16* __restrict__ w2f,
    const float* __restrict__ alphas,
    float* __restrict__ out,
    int Brows)
{
    __shared__ __align__(32) _Float16 lds_h[8][2][16][64];   // [wave][hi/lo][M][N]

    const int tid   = threadIdx.x;
    const int wave  = tid >> 5;
    const int lane  = tid & 31;
    const int lhalf = lane & 15;
    const int hsel  = (lane >> 4) & 1;
    const int m_base = (blockIdx.x * 8 + wave) * 16;

    const float alpha1 = alphas[0];
    const float alpha2 = alphas[1];

    int myrow = m_base + lhalf;
    if (myrow >= Brows) myrow = Brows - 1;          // clamp (keeps EXEC full)
    const float* rowp = x + (size_t)myrow * 784;

    v8f acc[4] = {};

    for (int kt = 0; kt < K1_TILES; ++kt) {
        const int g0 = kt * 32 + hsel * 8;
        // 16 f32 per lane: two contiguous 8-float runs (K pad zone -> zeros).
        // Per row per kt this touches exactly 128 contiguous bytes -> full
        // cacheline utilization while streaming x once.
        float4 f0 = *(const float4*)(rowp + g0);
        float4 f1 = *(const float4*)(rowp + g0 + 4);
        float4 f2 = make_float4(0.f, 0.f, 0.f, 0.f);
        float4 f3 = f2;
        if (kt < K1_TILES - 1) {
            f2 = *(const float4*)(rowp + g0 + 16);
            f3 = *(const float4*)(rowp + g0 + 20);
            __builtin_prefetch(rowp + g0 + 32, 0, 0);   // next K tile
        }
        float fv[16] = { f0.x, f0.y, f0.z, f0.w,  f1.x, f1.y, f1.z, f1.w,
                         f2.x, f2.y, f2.z, f2.w,  f3.x, f3.y, f3.z, f3.w };
        v16h a_hi, a_lo;
        #pragma unroll
        for (int i = 0; i < 16; ++i) {
            _Float16 h = (_Float16)fv[i];
            a_hi[i] = h;
            a_lo[i] = (_Float16)(fv[i] - (float)h);
        }

        const v16h* bp = (const v16h*)w1f + (size_t)kt * (N1_TILES * 32) + lane;
        #pragma unroll
        for (int nt = 0; nt < N1_TILES; ++nt) {
            v16h bf = bp[nt * 32];
            acc[nt] = __builtin_amdgcn_wmma_f32_16x16x32_f16(
                false, a_hi, false, bf, (short)0, acc[nt], false, false);
            acc[nt] = __builtin_amdgcn_wmma_f32_16x16x32_f16(
                false, a_lo, false, bf, (short)0, acc[nt], false, false);
        }
    }

    // relu(alpha1*acc + b1) -> LDS strip as hi/lo f16 (C layout: lane holds
    // N = nt*16 + lhalf, M = v + hsel*8)
    #pragma unroll
    for (int nt = 0; nt < N1_TILES; ++nt) {
        const int   N  = nt * 16 + lhalf;
        const float bn = b1[N];
        #pragma unroll
        for (int v = 0; v < 8; ++v) {
            const int M = v + hsel * 8;
            float hval = fmaxf(alpha1 * acc[nt][v] + bn, 0.f);
            _Float16 hh = (_Float16)hval;
            lds_h[wave][0][M][N] = hh;
            lds_h[wave][1][M][N] = (_Float16)(hval - (float)hh);
        }
    }
    __syncthreads();

    // Layer 2: h[16x64] @ Wq2^T (N padded to 16), K = 64 -> 2 WMMA K-tiles
    v8f oacc = {};
    #pragma unroll
    for (int k2 = 0; k2 < 2; ++k2) {
        const v16h bf = *((const v16h*)w2f + k2 * 32 + lane);
        const int kb = k2 * 32 + hsel * 8;
        const int M  = lhalf;
        v8h p0h = *(const v8h*)&lds_h[wave][0][M][kb];
        v8h p1h = *(const v8h*)&lds_h[wave][0][M][kb + 16];
        v8h p0l = *(const v8h*)&lds_h[wave][1][M][kb];
        v8h p1l = *(const v8h*)&lds_h[wave][1][M][kb + 16];
        v16h a2h = __builtin_shufflevector(p0h, p1h, 0,1,2,3,4,5,6,7,8,9,10,11,12,13,14,15);
        v16h a2l = __builtin_shufflevector(p0l, p1l, 0,1,2,3,4,5,6,7,8,9,10,11,12,13,14,15);
        oacc = __builtin_amdgcn_wmma_f32_16x16x32_f16(
            false, a2h, false, bf, (short)0, oacc, false, false);
        oacc = __builtin_amdgcn_wmma_f32_16x16x32_f16(
            false, a2l, false, bf, (short)0, oacc, false, false);
    }

    // Store out[M][j], j = lhalf (only j < 10 valid)
    const int j = lhalf;
    if (j < 10) {
        const float bj = b2[j];
        #pragma unroll
        for (int v = 0; v < 8; ++v) {
            const int M = m_base + v + hsel * 8;
            if (M < Brows)
                out[(size_t)M * 10 + j] = alpha2 * oacc[v] + bj;
        }
    }
}

// ---------------------------------------------------------------------------
extern "C" void kernel_launch(void* const* d_in, const int* in_sizes, int n_in,
                              void* d_out, int out_size, void* d_ws, size_t ws_size,
                              hipStream_t stream) {
    const float* x  = (const float*)d_in[0];
    const float* W1 = (const float*)d_in[1];
    const float* b1 = (const float*)d_in[2];
    const float* W2 = (const float*)d_in[3];
    const float* b2 = (const float*)d_in[4];
    float* out = (float*)d_out;

    char* ws = (char*)d_ws;
    float*    alphas = (float*)ws;                                  // 2 f32
    float*    part   = (float*)(ws + 64);                           // 384 f32
    _Float16* w1f    = (_Float16*)(ws + 2048);                      // 51200 halves
    _Float16* w2f    = (_Float16*)(ws + 2048 + W1FRAG_HALVES * 2);  // 1024 halves

    twn_abs_partial   <<<QBLOCKS, QTHREADS, 0, stream>>>(W1, W2, part);
    twn_masked_partial<<<QBLOCKS, QTHREADS, 0, stream>>>(W1, W2, part);
    twn_quant_write   <<<QBLOCKS, QTHREADS, 0, stream>>>(W1, W2, w1f, w2f, part, alphas);

    const int Brows  = in_sizes[0] / 784;          // 65536
    const int blocks = (Brows + 127) / 128;        // 8 waves x 16 rows per block
    twn_mlp_kernel<<<blocks, 256, 0, stream>>>(x, b1, b2, w1f, w2f, alphas, out, Brows);
}
